// DIFMultiHeadAttention_83966610636995
// MI455X (gfx1250) — matrix operand order repaired and checked by reading, not compiled, not run on hardware
//
#include <hip/hip_runtime.h>
#include <hip/hip_bf16.h>

typedef __attribute__((ext_vector_type(16))) __bf16 v16bf;
typedef __attribute__((ext_vector_type(8)))  __bf16 v8bf;
typedef __attribute__((ext_vector_type(8)))  float  v8f;
typedef __bf16 bf16;

#define SHUF16(lo, hi) __builtin_shufflevector((lo), (hi), 0,1,2,3,4,5,6,7,8,9,10,11,12,13,14,15)

static __device__ __forceinline__ bf16 f2bf(float f) {
  unsigned u = __builtin_bit_cast(unsigned, f);
  u += 0x7fffu + ((u >> 16) & 1u);  // round-to-nearest-even
  unsigned short h = (unsigned short)(u >> 16);
  return __builtin_bit_cast(bf16, h);
}

// LDS byte offset of a shared-memory pointer (flat shared addr low 32 bits == LDS offset)
static __device__ __forceinline__ unsigned ldsoff(const void* p) {
  return (unsigned)(unsigned long long)p;
}
// gfx1250 async DMA: 16 bytes global -> LDS, tracked by ASYNCcnt
static __device__ __forceinline__ void async_copy16(unsigned lds, const void* gptr) {
  asm volatile("global_load_async_to_lds_b128 %0, %1, off"
               :: "v"(lds), "v"(gptr) : "memory");
}
template <int N>
static __device__ __forceinline__ void wait_async() {
  asm volatile("s_wait_asynccnt %0" :: "i"(N) : "memory");
}

// B=32, S=512, D=512, H=8, HD=64, DA=256, HA=32, KC=192, NTOK=16384
#define S_LEN 512
#define NTOK  16384
#define KC    192

// ---------------------------------------------------------------- converts
__global__ __launch_bounds__(256) void cvt_f32_bf16(const float* __restrict__ src,
                                                    bf16* __restrict__ dst, int n) {
  for (int i = blockIdx.x * 256 + threadIdx.x; i < n; i += gridDim.x * 256)
    dst[i] = f2bf(src[i]);
}

// ---------------------------------------------------------------- projection GEMM
// C[n,o] = sum_k A[n,k]*W[o,k] + bias[o]  (W is bf16 (out,in) row-major)
// vtrans==0 : dst = ((b*nheads+h)*512 + s)*dstride + (o%headw) + coff
// vtrans==1 : dst = ((b*nheads+h)*headw + (o%headw))*512 + s      (V transposed)
__global__ __launch_bounds__(256)
void gemm_bf16(const bf16* __restrict__ A, const bf16* __restrict__ W,
               const float* __restrict__ bias, bf16* __restrict__ Cout,
               int Kdim, int Odim, int headw, int dstride, int coff, int vtrans) {
  __shared__ bf16 As[2][128][40];
  __shared__ bf16 Bs[2][128][40];
  const int t = threadIdx.x, lane = t & 31, wave = t >> 5;
  const int wm = wave >> 1, wn = wave & 1;
  const int row0 = blockIdx.y * 128, o0 = blockIdx.x * 128;
  const int mlo = lane & 15, hi = lane >> 4;
  const int khalf = hi * 8, kbase = hi * 16;

  v8f acc[2][4];
#pragma unroll
  for (int mi = 0; mi < 2; ++mi)
#pragma unroll
    for (int ni = 0; ni < 4; ++ni) acc[mi][ni] = (v8f){0, 0, 0, 0, 0, 0, 0, 0};

  // Stage one 128x32 A tile + 128x32 B tile: 8 async 16B copies per thread.
  auto stage = [&](int buf, int k0) {
#pragma unroll
    for (int j = 0; j < 4; ++j) {
      int ci = t + j * 256;             // 1024 chunks of 8 halves
      int r = ci >> 2, sub = (ci & 3) * 8;
      async_copy16(ldsoff(&As[buf][r][sub]),
                   A + (size_t)(row0 + r) * Kdim + k0 + sub);
      async_copy16(ldsoff(&Bs[buf][r][sub]),
                   W + (size_t)(o0 + r) * Kdim + k0 + sub);
    }
  };

  const int nt = Kdim >> 5;
  stage(0, 0);
  for (int i = 0; i < nt; ++i) {
    const int buf = i & 1;
    __syncthreads();                    // prior reads of buf^1 complete
    if (i + 1 < nt) {
      stage(buf ^ 1, (i + 1) * 32);     // overlap next DMA with this tile's math
      wait_async<8>();                  // oldest 8 (tile i) complete, newest 8 in flight
    } else {
      wait_async<0>();
    }
    __syncthreads();                    // tile i visible to all waves

    v16bf af[2], bfr[4];
#pragma unroll
    for (int mi = 0; mi < 2; ++mi) {
      const bf16* p = &As[buf][wm * 32 + mi * 16 + mlo][0];
      af[mi] = SHUF16(*(const v8bf*)(p + khalf), *(const v8bf*)(p + 16 + khalf));
    }
#pragma unroll
    for (int ni = 0; ni < 4; ++ni) {
      const bf16* p = &Bs[buf][wn * 64 + ni * 16 + mlo][0];
      bfr[ni] = SHUF16(*(const v8bf*)(p + khalf), *(const v8bf*)(p + 16 + khalf));
    }
#pragma unroll
    for (int mi = 0; mi < 2; ++mi)
#pragma unroll
      for (int ni = 0; ni < 4; ++ni)
        acc[mi][ni] = __builtin_amdgcn_wmma_f32_16x16x32_bf16(
            false, af[mi], false, bfr[ni], (short)0, acc[mi][ni], false, false);
  }

  const int nheads = Odim / headw;
#pragma unroll
  for (int mi = 0; mi < 2; ++mi)
#pragma unroll
    for (int ni = 0; ni < 4; ++ni)
#pragma unroll
      for (int r = 0; r < 8; ++r) {
        int gm = row0 + wm * 32 + mi * 16 + hi * 8 + r;
        int go = o0 + wn * 64 + ni * 16 + mlo;
        float v = acc[mi][ni][r] + bias[go];
        int bb = gm >> 9, ss = gm & 511;
        int hh = go / headw, cc = go - hh * headw;
        size_t dst;
        if (vtrans)
          dst = (((size_t)(bb * nheads + hh) * headw + cc) << 9) + (size_t)ss;
        else
          dst = ((size_t)(bb * nheads + hh) * S_LEN + ss) * (size_t)dstride + cc + coff;
        Cout[dst] = f2bf(v);
      }
}

// ---------------------------------------------------------------- final GEMM (+bias+residual, f32 out)
__global__ __launch_bounds__(256)
void gemm_final(const bf16* __restrict__ A, const bf16* __restrict__ W,
                const float* __restrict__ bias, const float* __restrict__ resid,
                float* __restrict__ Out) {
  __shared__ bf16 As[2][128][40];
  __shared__ bf16 Bs[2][128][40];
  const int t = threadIdx.x, lane = t & 31, wave = t >> 5;
  const int wm = wave >> 1, wn = wave & 1;
  const int row0 = blockIdx.y * 128, o0 = blockIdx.x * 128;
  const int mlo = lane & 15, hi = lane >> 4;
  const int khalf = hi * 8, kbase = hi * 16;

  v8f acc[2][4];
#pragma unroll
  for (int mi = 0; mi < 2; ++mi)
#pragma unroll
    for (int ni = 0; ni < 4; ++ni) acc[mi][ni] = (v8f){0, 0, 0, 0, 0, 0, 0, 0};

  auto stage = [&](int buf, int k0) {
#pragma unroll
    for (int j = 0; j < 4; ++j) {
      int ci = t + j * 256;
      int r = ci >> 2, sub = (ci & 3) * 8;
      async_copy16(ldsoff(&As[buf][r][sub]), A + (size_t)(row0 + r) * 512 + k0 + sub);
      async_copy16(ldsoff(&Bs[buf][r][sub]), W + (size_t)(o0 + r) * 512 + k0 + sub);
    }
  };

  stage(0, 0);
  for (int i = 0; i < 16; ++i) {
    const int buf = i & 1;
    __syncthreads();
    if (i + 1 < 16) {
      stage(buf ^ 1, (i + 1) * 32);
      wait_async<8>();
    } else {
      wait_async<0>();
    }
    __syncthreads();

    v16bf af[2], bfr[4];
#pragma unroll
    for (int mi = 0; mi < 2; ++mi) {
      const bf16* p = &As[buf][wm * 32 + mi * 16 + mlo][0];
      af[mi] = SHUF16(*(const v8bf*)(p + khalf), *(const v8bf*)(p + 16 + khalf));
    }
#pragma unroll
    for (int ni = 0; ni < 4; ++ni) {
      const bf16* p = &Bs[buf][wn * 64 + ni * 16 + mlo][0];
      bfr[ni] = SHUF16(*(const v8bf*)(p + khalf), *(const v8bf*)(p + 16 + khalf));
    }
#pragma unroll
    for (int mi = 0; mi < 2; ++mi)
#pragma unroll
      for (int ni = 0; ni < 4; ++ni)
        acc[mi][ni] = __builtin_amdgcn_wmma_f32_16x16x32_bf16(
            false, af[mi], false, bfr[ni], (short)0, acc[mi][ni], false, false);
  }

#pragma unroll
  for (int mi = 0; mi < 2; ++mi)
#pragma unroll
    for (int ni = 0; ni < 4; ++ni)
#pragma unroll
      for (int r = 0; r < 8; ++r) {
        int gm = row0 + wm * 32 + mi * 16 + hi * 8 + r;
        int go = o0 + wn * 64 + ni * 16 + mlo;
        size_t idx = (size_t)gm * 512 + go;
        Out[idx] = acc[mi][ni][r] + bias[go] + resid[idx];
      }
}

// ---------------------------------------------------------------- flash attention
// Qc,Kc: (b,h,s,192) bf16;  Vt: (b,h,64,S) bf16;  mask: (b,S) f32;  ctx: (token,512) bf16
__global__ __launch_bounds__(256)
void attn_kernel(const bf16* __restrict__ Qc, const bf16* __restrict__ Kc,
                 const bf16* __restrict__ Vt, const float* __restrict__ mask,
                 bf16* __restrict__ ctx) {
  __shared__ bf16 Ks[32][200];       // 32 keys x 192 (padded)
  __shared__ bf16 Vs[64][40];        // 64 hd x 32 keys (padded)
  __shared__ bf16 plds[8][16][40];   // per-wave P tile
  const int t = threadIdx.x, lane = t & 31, wave = t >> 5;
  const int b = blockIdx.z, h = blockIdx.y;
  const int q0 = blockIdx.x * 128 + wave * 16;
  const size_t bh = (size_t)(b * 8 + h);
  const bf16* Qrow = Qc + (bh * S_LEN + q0) * KC;
  const bf16* Krow = Kc + bh * S_LEN * KC;
  const bf16* Vrow = Vt + bh * 64 * S_LEN;
  const float* mrow = mask + (size_t)b * S_LEN;
  const int mlo = lane & 15, hi = lane >> 4;
  const int khalf = hi * 8, kbase = hi * 16;

  v16bf aq[6];
#pragma unroll
  for (int c = 0; c < 6; ++c) {
    const bf16* p = Qrow + (size_t)mlo * KC + c * 32;
    aq[c] = SHUF16(*(const v8bf*)(p + khalf), *(const v8bf*)(p + 16 + khalf));
  }

  float m_i[8], l_i[8];
  v8f o_acc[4];
#pragma unroll
  for (int r = 0; r < 8; ++r) { m_i[r] = -3.0e38f; l_i[r] = 0.0f; }
#pragma unroll
  for (int ni = 0; ni < 4; ++ni) o_acc[ni] = (v8f){0, 0, 0, 0, 0, 0, 0, 0};

  for (int kt2 = 0; kt2 < 16; ++kt2) {  // 32 keys per iteration
    __syncthreads();                    // previous tile's LDS reads done
    {                                   // K tile: 768 chunks (3/thread), V tile: 256 chunks (1/thread)
#pragma unroll
      for (int j = 0; j < 3; ++j) {
        int ci = t + j * 256;
        int r = ci / 24, sub = (ci - r * 24) * 8;
        async_copy16(ldsoff(&Ks[r][sub]),
                     Krow + (size_t)(kt2 * 32 + r) * KC + sub);
      }
      int vr = t >> 2, vp = (t & 3) * 8;
      async_copy16(ldsoff(&Vs[vr][vp]),
                   Vrow + (size_t)vr * S_LEN + kt2 * 32 + vp);
      wait_async<0>();
    }
    __syncthreads();

    v8f s0 = (v8f){0, 0, 0, 0, 0, 0, 0, 0};
    v8f s1 = (v8f){0, 0, 0, 0, 0, 0, 0, 0};
#pragma unroll
    for (int c = 0; c < 6; ++c) {
      const bf16* p0 = &Ks[mlo][c * 32 + khalf];
      v16bf bk0 = SHUF16(*(const v8bf*)(p0), *(const v8bf*)(p0 + 16));
      s0 = __builtin_amdgcn_wmma_f32_16x16x32_bf16(false, aq[c], false, bk0,
                                                   (short)0, s0, false, false);
      const bf16* p1 = &Ks[16 + mlo][c * 32 + khalf];
      v16bf bk1 = SHUF16(*(const v8bf*)(p1), *(const v8bf*)(p1 + 16));
      s1 = __builtin_amdgcn_wmma_f32_16x16x32_bf16(false, aq[c], false, bk1,
                                                   (short)0, s1, false, false);
    }

    const float mv0 = mrow[kt2 * 32 + mlo];
    const float mv1 = mrow[kt2 * 32 + 16 + mlo];
    float ef[8];
#pragma unroll
    for (int r = 0; r < 8; ++r) {
      float v0 = s0[r] * 0.125f + mv0;  // 1/SCALE = 1/8
      float v1 = s1[r] * 0.125f + mv1;
      float tmax = fmaxf(v0, v1);
      tmax = fmaxf(tmax, __shfl_xor(tmax, 1));
      tmax = fmaxf(tmax, __shfl_xor(tmax, 2));
      tmax = fmaxf(tmax, __shfl_xor(tmax, 4));
      tmax = fmaxf(tmax, __shfl_xor(tmax, 8));
      float mn = fmaxf(m_i[r], tmax);
      float e = __expf(m_i[r] - mn);
      float p0 = __expf(v0 - mn);
      float p1 = __expf(v1 - mn);
      float rs = p0 + p1;
      rs += __shfl_xor(rs, 1);
      rs += __shfl_xor(rs, 2);
      rs += __shfl_xor(rs, 4);
      rs += __shfl_xor(rs, 8);
      l_i[r] = l_i[r] * e + rs;
      m_i[r] = mn;
      ef[r] = e;
      plds[wave][hi * 8 + r][mlo] = f2bf(p0);
      plds[wave][hi * 8 + r][16 + mlo] = f2bf(p1);
    }
#pragma unroll
    for (int ni = 0; ni < 4; ++ni)
#pragma unroll
      for (int r = 0; r < 8; ++r) o_acc[ni][r] *= ef[r];

    const bf16* pp = &plds[wave][mlo][0];
    v16bf ap = SHUF16(*(const v8bf*)(pp + khalf), *(const v8bf*)(pp + 16 + khalf));
#pragma unroll
    for (int ni = 0; ni < 4; ++ni) {
      const bf16* vp = &Vs[ni * 16 + mlo][khalf];
      v16bf bv = SHUF16(*(const v8bf*)(vp), *(const v8bf*)(vp + 16));
      o_acc[ni] = __builtin_amdgcn_wmma_f32_16x16x32_bf16(false, ap, false, bv,
                                                          (short)0, o_acc[ni], false, false);
    }
  }

#pragma unroll
  for (int r = 0; r < 8; ++r) {
    float inv = 1.0f / l_i[r];
    int srow = q0 + hi * 8 + r;
    size_t base = ((size_t)(b * S_LEN + srow)) * 512 + h * 64;
#pragma unroll
    for (int ni = 0; ni < 4; ++ni)
      ctx[base + ni * 16 + mlo] = f2bf(o_acc[ni][r] * inv);
  }
}

// ---------------------------------------------------------------- in-place LayerNorm
__global__ __launch_bounds__(256)
void ln_kernel(float* __restrict__ Out, const float* __restrict__ gamma,
               const float* __restrict__ beta) {
  __shared__ float ssum[8], ssq[8];
  const int t = threadIdx.x, lane = t & 31, wave = t >> 5;
  const size_t row = (size_t)blockIdx.x * 512;
  float x0 = Out[row + t], x1 = Out[row + 256 + t];
  float s = x0 + x1, q = x0 * x0 + x1 * x1;
#pragma unroll
  for (int off = 1; off < 32; off <<= 1) {
    s += __shfl_xor(s, off);
    q += __shfl_xor(q, off);
  }
  if (lane == 0) { ssum[wave] = s; ssq[wave] = q; }
  __syncthreads();
  float S = 0.0f, Q = 0.0f;
#pragma unroll
  for (int w = 0; w < 8; ++w) { S += ssum[w]; Q += ssq[w]; }
  float mu = S * (1.0f / 512.0f);
  float var = Q * (1.0f / 512.0f) - mu * mu;
  float rinv = rsqrtf(var + 1e-5f);
  Out[row + t] = (x0 - mu) * rinv * gamma[t] + beta[t];
  Out[row + 256 + t] = (x1 - mu) * rinv * gamma[256 + t] + beta[256 + t];
}

// ---------------------------------------------------------------- launcher
extern "C" void kernel_launch(void* const* d_in, const int* in_sizes, int n_in,
                              void* d_out, int out_size, void* d_ws, size_t ws_size,
                              hipStream_t stream) {
  const float* item = (const float*)d_in[0];
  const float* attr0 = (const float*)d_in[1];
  const float* attr1 = (const float*)d_in[2];
  const float* pos = (const float*)d_in[3];
  const float* amask = (const float*)d_in[4];
  const float* Wq = (const float*)d_in[5];   const float* bq = (const float*)d_in[6];
  const float* Wk = (const float*)d_in[7];   const float* bk = (const float*)d_in[8];
  const float* Wv = (const float*)d_in[9];   const float* bv = (const float*)d_in[10];
  const float* Wqp = (const float*)d_in[11]; const float* bqp = (const float*)d_in[12];
  const float* Wkp = (const float*)d_in[13]; const float* bkp = (const float*)d_in[14];
  const float* Wqa0 = (const float*)d_in[15]; const float* bqa0 = (const float*)d_in[16];
  const float* Wka0 = (const float*)d_in[17]; const float* bka0 = (const float*)d_in[18];
  const float* Wqa1 = (const float*)d_in[19]; const float* bqa1 = (const float*)d_in[20];
  const float* Wka1 = (const float*)d_in[21]; const float* bka1 = (const float*)d_in[22];
  const float* Wd = (const float*)d_in[23];  const float* bd = (const float*)d_in[24];
  const float* gamma = (const float*)d_in[25];
  const float* beta = (const float*)d_in[26];
  float* Out = (float*)d_out;

  size_t off = 0;
  auto alloc = [&](size_t bytes) -> char* {
    char* p = (char*)d_ws + off;
    off = (off + bytes + 255) & ~(size_t)255;
    return p;
  };
  bf16* Xi  = (bf16*)alloc((size_t)NTOK * 512 * 2);
  bf16* Xp  = (bf16*)alloc((size_t)NTOK * 512 * 2);
  bf16* Xa0 = (bf16*)alloc((size_t)NTOK * 256 * 2);
  bf16* Xa1 = (bf16*)alloc((size_t)NTOK * 256 * 2);
  bf16* WbQ  = (bf16*)alloc(512 * 512 * 2);
  bf16* WbK  = (bf16*)alloc(512 * 512 * 2);
  bf16* WbV  = (bf16*)alloc(512 * 512 * 2);
  bf16* WbQp = (bf16*)alloc(512 * 512 * 2);
  bf16* WbKp = (bf16*)alloc(512 * 512 * 2);
  bf16* WbQa0 = (bf16*)alloc(256 * 256 * 2);
  bf16* WbKa0 = (bf16*)alloc(256 * 256 * 2);
  bf16* WbQa1 = (bf16*)alloc(256 * 256 * 2);
  bf16* WbKa1 = (bf16*)alloc(256 * 256 * 2);
  bf16* WbD  = (bf16*)alloc(512 * 512 * 2);
  bf16* Qcat = (bf16*)alloc((size_t)32 * 8 * 512 * KC * 2);
  bf16* Kcat = (bf16*)alloc((size_t)32 * 8 * 512 * KC * 2);
  bf16* Vtr  = (bf16*)alloc((size_t)32 * 8 * 64 * 512 * 2);
  bf16* Ctx  = (bf16*)alloc((size_t)NTOK * 512 * 2);
  (void)ws_size; (void)in_sizes; (void)n_in; (void)out_size;

  const int CB = 2048;
  cvt_f32_bf16<<<CB, 256, 0, stream>>>(item, Xi, NTOK * 512);
  cvt_f32_bf16<<<CB, 256, 0, stream>>>(pos, Xp, NTOK * 512);
  cvt_f32_bf16<<<CB, 256, 0, stream>>>(attr0, Xa0, NTOK * 256);
  cvt_f32_bf16<<<CB, 256, 0, stream>>>(attr1, Xa1, NTOK * 256);
  cvt_f32_bf16<<<256, 256, 0, stream>>>(Wq, WbQ, 512 * 512);
  cvt_f32_bf16<<<256, 256, 0, stream>>>(Wk, WbK, 512 * 512);
  cvt_f32_bf16<<<256, 256, 0, stream>>>(Wv, WbV, 512 * 512);
  cvt_f32_bf16<<<256, 256, 0, stream>>>(Wqp, WbQp, 512 * 512);
  cvt_f32_bf16<<<256, 256, 0, stream>>>(Wkp, WbKp, 512 * 512);
  cvt_f32_bf16<<<64, 256, 0, stream>>>(Wqa0, WbQa0, 256 * 256);
  cvt_f32_bf16<<<64, 256, 0, stream>>>(Wka0, WbKa0, 256 * 256);
  cvt_f32_bf16<<<64, 256, 0, stream>>>(Wqa1, WbQa1, 256 * 256);
  cvt_f32_bf16<<<64, 256, 0, stream>>>(Wka1, WbKa1, 256 * 256);
  cvt_f32_bf16<<<256, 256, 0, stream>>>(Wd, WbD, 512 * 512);

  dim3 gD(4, 128);   // Odim 512 projections
  dim3 gA(2, 128);   // Odim 256 projections
  gemm_bf16<<<gD, 256, 0, stream>>>(Xi, WbQ, bq, Qcat, 512, 512, 64, KC, 0, 0);
  gemm_bf16<<<gD, 256, 0, stream>>>(Xi, WbK, bk, Kcat, 512, 512, 64, KC, 0, 0);
  gemm_bf16<<<gD, 256, 0, stream>>>(Xp, WbQp, bqp, Qcat, 512, 512, 64, KC, 64, 0);
  gemm_bf16<<<gD, 256, 0, stream>>>(Xp, WbKp, bkp, Kcat, 512, 512, 64, KC, 64, 0);
  gemm_bf16<<<gA, 256, 0, stream>>>(Xa0, WbQa0, bqa0, Qcat, 256, 256, 32, KC, 128, 0);
  gemm_bf16<<<gA, 256, 0, stream>>>(Xa0, WbKa0, bka0, Kcat, 256, 256, 32, KC, 128, 0);
  gemm_bf16<<<gA, 256, 0, stream>>>(Xa1, WbQa1, bqa1, Qcat, 256, 256, 32, KC, 160, 0);
  gemm_bf16<<<gA, 256, 0, stream>>>(Xa1, WbKa1, bka1, Kcat, 256, 256, 32, KC, 160, 0);
  gemm_bf16<<<gD, 256, 0, stream>>>(Xi, WbV, bv, Vtr, 512, 512, 64, 0, 0, 1);

  attn_kernel<<<dim3(4, 8, 32), 256, 0, stream>>>(Qcat, Kcat, Vtr, amask, Ctx);

  gemm_final<<<gD, 256, 0, stream>>>(Ctx, WbD, bd, item, Out);
  ln_kernel<<<NTOK, 256, 0, stream>>>(Out, gamma, beta);
}
